// PositionalTransformer_32341103739540
// MI455X (gfx1250) — matrix operand order repaired and checked by reading, not compiled, not run on hardware
//
#include <hip/hip_runtime.h>
#include <hip/hip_bf16.h>

typedef __attribute__((ext_vector_type(16))) __bf16 v16bf;
typedef __attribute__((ext_vector_type(8)))  __bf16 v8bf;
typedef __attribute__((ext_vector_type(8)))  float  v8f;

#define DMODEL   512
#define WWIN     8
#define NPOS     9            // WWIN + 1
#define NLAYERS  3
#define TILE_M   16
#define THREADS  512          // 16 waves of 32
#define MAT_ELEMS (512*512)   // 262144 = 2^18
#define NMAT     67           // 3 Wq + 27 Wk + 24 Wfw + 3 Wfc + 1 Wqf + 9 Wkf
#define EPSV     1e-12f

// ---------------------------------------------------------------------------
// Pre-pass: convert fp32 weights -> bf16 and swizzle into WMMA-B fragment
// order.  Fragment (ntile, ktile) of a 512x512 matrix is a 32x16 (KxN) tile;
// packed layout: [matrix][frag = ntile*16+ktile][lane][j], 16 bf16 per lane
// (one contiguous 32B chunk -> one coalesced v16bf load in the GEMM).
// B[k][n] = W[n][k]  (every matmul in the reference is  x @ W^T).
// Lane mapping: n = lane&15,  k = ktile*32 + (lane>>4)*16 + j.
// ---------------------------------------------------------------------------
__global__ __launch_bounds__(256) void pack_weights_kernel(
    const float* __restrict__ Wq,  const float* __restrict__ Wk,
    const float* __restrict__ Wfw, const float* __restrict__ Wfc,
    const float* __restrict__ Wqf, const float* __restrict__ Wkf,
    __bf16* __restrict__ packed)
{
    long idx = (long)blockIdx.x * 256 + threadIdx.x;
    const long total = (long)NMAT * MAT_ELEMS;
    if (idx >= total) return;
    int m    = (int)(idx >> 18);
    int r    = (int)(idx & (MAT_ELEMS - 1));
    int f    = r >> 9;            // fragment id
    int lane = (r >> 4) & 31;
    int j    = r & 15;
    int n    = (f >> 4) * 16 + (lane & 15);
    int k    = (f & 15) * 32 + (lane >> 4) * 16 + j;
    const float* src;
    if      (m < 3)   src = Wq  + (long)m * MAT_ELEMS;
    else if (m < 30)  src = Wk  + (long)(m - 3)  * MAT_ELEMS;
    else if (m < 54)  src = Wfw + (long)(m - 30) * MAT_ELEMS;
    else if (m < 57)  src = Wfc + (long)(m - 54) * MAT_ELEMS;
    else if (m == 57) src = Wqf;
    else              src = Wkf + (long)(m - 58) * MAT_ELEMS;
    packed[idx] = (__bf16)src[n * DMODEL + k];
}

// ---------------------------------------------------------------------------
// One wave computes a 16x32 slab of  D = A(16x512,bf16) * B(512x512,bf16) + b
// (16 waves cover all 512 columns).  A from LDS; B streamed from the packed
// global weight image; fp32 result (+bias) written to a 16x512 LDS buffer.
// While streaming, prefetch the next GEMM's fragments (global_prefetch_b8)
// so the L2-resident weights are already in near cache when needed.
// ---------------------------------------------------------------------------
__device__ __forceinline__ void gemm_tile(
    const __bf16* __restrict__ aBase, int aStride,
    const __bf16* __restrict__ bMat,
    const __bf16* __restrict__ bNext,   // next weight matrix (may be null)
    const float*  __restrict__ bias,
    float* __restrict__ outF,
    int wid, int lane)
{
    v8f c[2] = {};
    const int m   = lane & 15;
    const int sel = lane >> 4;
    const int nt0 = wid * 2;
    union { v16bf v; v8bf h[2]; } a;
    #pragma unroll 4
    for (int kt = 0; kt < 16; ++kt) {
        const __bf16* ap = aBase + m * aStride + kt * 32 + sel * 16;
        a.h[0] = *(const v8bf*)(ap);
        a.h[1] = *(const v8bf*)(ap + 8);
        #pragma unroll
        for (int t = 0; t < 2; ++t) {
            const long boff = ((((nt0 + t) * 16 + kt)) << 9) + (lane << 4);
            const v16bf b = *(const v16bf*)(bMat + boff);
            if (bNext) __builtin_prefetch(bNext + boff, 0, 1);
            c[t] = __builtin_amdgcn_wmma_f32_16x16x32_bf16(
                       false, a.v, false, b, (short)0, c[t], false, false);
        }
    }
    #pragma unroll
    for (int t = 0; t < 2; ++t) {
        const int ncol = (nt0 + t) * 16 + m;
        const float bv = bias[ncol];
        #pragma unroll
        for (int r = 0; r < 8; ++r)
            outF[(r + sel * 8) * DMODEL + ncol] = c[t][r] + bv;
    }
}

// ---------------------------------------------------------------------------
// Fully fused network: 16 batch rows per workgroup, everything on-chip.
// ---------------------------------------------------------------------------
__global__ __launch_bounds__(THREADS) void pt_fused_kernel(
    const float* __restrict__ x,
    const float* __restrict__ bq,   // [3][512]
    const float* __restrict__ bk,   // [3][9][512]
    const float* __restrict__ bfw,  // [3][8][512]
    const float* __restrict__ bfc,  // [3][512]
    const float* __restrict__ bqf,  // [512]
    const float* __restrict__ bkf,  // [9][512]
    const __bf16* __restrict__ wp,  // packed bf16 weights
    float* __restrict__ out)
{
    extern __shared__ char smem[];
    __bf16* sx    = (__bf16*)smem;                               // [16][9][512] bf16
    float*  sq    = (float*) (smem + 147456);                    // [16][512]
    float*  skb   = (float*) (smem + 147456 + 32768);            // [16][512] scratch
    __bf16* sc    = (__bf16*)sq;                                 // alias: q dead after scores
    float*  sred  = (float*) (smem + 147456 + 65536);            // [16]
    float*  sattn = sred + 16;                                   // [16][9]

    const int tid  = threadIdx.x;
    const int wid  = tid >> 5;
    const int lane = tid & 31;
    const long b0  = (long)blockIdx.x * TILE_M;

    // matrix bases in the packed image
    #define WMAT(m_) (wp + (long)(m_) * MAT_ELEMS)
    // m-indices: Wq[l]=l  Wk[l][p]=3+l*9+p  Wfw[l][p]=30+l*8+p  Wfc[l]=54+l
    //            Wqf=57   Wkf[p]=58+p

    // ---- load x tile, fp32 -> bf16 in LDS ----
    const float* xg = x + b0 * (NPOS * DMODEL);
    for (int i = tid; i < TILE_M * NPOS * DMODEL; i += THREADS)
        sx[i] = (__bf16)xg[i];
    __syncthreads();

    const int row = tid >> 5;        // 0..15
    const int d0  = (tid & 31) * 16; // 16 elements per thread

    for (int l = 0; l < NLAYERS; ++l) {
        // q = xc @ Wq^T + bq
        gemm_tile(sx + 8 * DMODEL, NPOS * DMODEL,
                  WMAT(l), WMAT(3 + l * 9), bq + l * DMODEL, sq, wid, lane);
        if (tid < TILE_M * NPOS) sattn[tid] = 0.f;
        __syncthreads();

        // scores: s[b][p] = q . (x_p @ Wk_p^T + bk_p)
        for (int p = 0; p < NPOS; ++p) {
            const __bf16* nxt = (p < NPOS - 1) ? WMAT(3 + l * 9 + p + 1) : WMAT(54 + l);
            gemm_tile(sx + p * DMODEL, NPOS * DMODEL,
                      WMAT(3 + l * 9 + p), nxt,
                      bk + (l * 9 + p) * DMODEL, skb, wid, lane);
            __syncthreads();
            float part = 0.f;
            for (int d = d0; d < d0 + 16; ++d)
                part += sq[row * DMODEL + d] * skb[row * DMODEL + d];
            atomicAdd(&sattn[row * NPOS + p], part);
            __syncthreads();
        }

        // softmax over 9 positions (one thread per row)
        if (tid < TILE_M) {
            float mx = -3.4e38f;
            for (int p = 0; p < NPOS; ++p) mx = fmaxf(mx, sattn[tid * NPOS + p]);
            float e[NPOS], s = 0.f;
            for (int p = 0; p < NPOS; ++p) { e[p] = __expf(sattn[tid * NPOS + p] - mx); s += e[p]; }
            const float inv = 1.f / s;
            for (int p = 0; p < NPOS; ++p) sattn[tid * NPOS + p] = e[p] * inv;
            sred[tid] = 0.f;
        }
        __syncthreads();

        // x_cent = sum_p attn*x ; l2norm -> bf16 stage (sc aliases sq)
        {
            float ss = 0.f;
            for (int d = d0; d < d0 + 16; ++d) {
                float c = 0.f;
                for (int p = 0; p < NPOS; ++p)
                    c += sattn[row * NPOS + p] * (float)sx[(row * NPOS + p) * DMODEL + d];
                skb[row * DMODEL + d] = c;
                ss += c * c;
            }
            atomicAdd(&sred[row], ss);
        }
        __syncthreads();
        {
            const float scale = 1.f / fmaxf(sqrtf(sred[row]), EPSV);
            for (int d = d0; d < d0 + 16; ++d)
                sc[row * DMODEL + d] = (__bf16)(skb[row * DMODEL + d] * scale);
        }
        __syncthreads();

        // fc projection
        gemm_tile(sc, DMODEL, WMAT(54 + l), WMAT(30 + l * 8),
                  bfc + l * DMODEL, skb, wid, lane);
        if (tid < TILE_M) sred[tid] = 0.f;
        __syncthreads();

        // residual with xc + l2norm -> new center row
        {
            float ss = 0.f;
            for (int d = d0; d < d0 + 16; ++d) {
                float t = skb[row * DMODEL + d] + (float)sx[(row * NPOS + 8) * DMODEL + d];
                skb[row * DMODEL + d] = t;
                ss += t * t;
            }
            atomicAdd(&sred[row], ss);
        }
        __syncthreads();
        {
            const float scale = 1.f / fmaxf(sqrtf(sred[row]), EPSV);
            for (int d = d0; d < d0 + 16; ++d)
                sx[(row * NPOS + 8) * DMODEL + d] = (__bf16)(skb[row * DMODEL + d] * scale);
        }
        __syncthreads();

        // window positions: x_p = l2norm(x_p @ Wfw_p^T + bfw_p + x_p)
        for (int p = 0; p < WWIN; ++p) {
            const __bf16* nxt = (p < WWIN - 1) ? WMAT(30 + l * 8 + p + 1)
                               : ((l < NLAYERS - 1) ? WMAT(l + 1) : WMAT(57));
            gemm_tile(sx + p * DMODEL, NPOS * DMODEL,
                      WMAT(30 + l * 8 + p), nxt,
                      bfw + (l * 8 + p) * DMODEL, skb, wid, lane);
            if (tid < TILE_M) sred[tid] = 0.f;
            __syncthreads();
            float ss = 0.f;
            for (int d = d0; d < d0 + 16; ++d) {
                float t = skb[row * DMODEL + d] + (float)sx[(row * NPOS + p) * DMODEL + d];
                skb[row * DMODEL + d] = t;
                ss += t * t;
            }
            atomicAdd(&sred[row], ss);
            __syncthreads();
            const float scale = 1.f / fmaxf(sqrtf(sred[row]), EPSV);
            for (int d = d0; d < d0 + 16; ++d)
                sx[(row * NPOS + p) * DMODEL + d] = (__bf16)(skb[row * DMODEL + d] * scale);
            __syncthreads();
        }
    }

    // ---- final head: attn_cent with Wqf/Wkf, out = l2norm(x_cent + xc) ----
    gemm_tile(sx + 8 * DMODEL, NPOS * DMODEL,
              WMAT(57), WMAT(58), bqf, sq, wid, lane);
    if (tid < TILE_M * NPOS) sattn[tid] = 0.f;
    __syncthreads();
    for (int p = 0; p < NPOS; ++p) {
        const __bf16* nxt = (p < NPOS - 1) ? WMAT(58 + p + 1) : nullptr;
        gemm_tile(sx + p * DMODEL, NPOS * DMODEL,
                  WMAT(58 + p), nxt, bkf + p * DMODEL, skb, wid, lane);
        __syncthreads();
        float part = 0.f;
        for (int d = d0; d < d0 + 16; ++d)
            part += sq[row * DMODEL + d] * skb[row * DMODEL + d];
        atomicAdd(&sattn[row * NPOS + p], part);
        __syncthreads();
    }
    if (tid < TILE_M) {
        float mx = -3.4e38f;
        for (int p = 0; p < NPOS; ++p) mx = fmaxf(mx, sattn[tid * NPOS + p]);
        float e[NPOS], s = 0.f;
        for (int p = 0; p < NPOS; ++p) { e[p] = __expf(sattn[tid * NPOS + p] - mx); s += e[p]; }
        const float inv = 1.f / s;
        for (int p = 0; p < NPOS; ++p) sattn[tid * NPOS + p] = e[p] * inv;
        sred[tid] = 0.f;
    }
    __syncthreads();
    {
        float ss = 0.f;
        for (int d = d0; d < d0 + 16; ++d) {
            float c = (float)sx[(row * NPOS + 8) * DMODEL + d];   // + xc
            for (int p = 0; p < NPOS; ++p)
                c += sattn[row * NPOS + p] * (float)sx[(row * NPOS + p) * DMODEL + d];
            skb[row * DMODEL + d] = c;
            ss += c * c;
        }
        atomicAdd(&sred[row], ss);
    }
    __syncthreads();
    {
        const float scale = 1.f / fmaxf(sqrtf(sred[row]), EPSV);
        for (int d = d0; d < d0 + 16; ++d)
            out[(b0 + row) * DMODEL + d] = skb[row * DMODEL + d] * scale;
    }
    #undef WMAT
}

// ---------------------------------------------------------------------------
extern "C" void kernel_launch(void* const* d_in, const int* in_sizes, int n_in,
                              void* d_out, int out_size, void* d_ws, size_t ws_size,
                              hipStream_t stream) {
    const float* x   = (const float*)d_in[0];
    const float* Wq  = (const float*)d_in[1];
    const float* bq  = (const float*)d_in[2];
    const float* Wk  = (const float*)d_in[3];
    const float* bk  = (const float*)d_in[4];
    const float* Wfw = (const float*)d_in[5];
    const float* bfw = (const float*)d_in[6];
    const float* Wfc = (const float*)d_in[7];
    const float* bfc = (const float*)d_in[8];
    const float* Wqf = (const float*)d_in[9];
    const float* bqf = (const float*)d_in[10];
    const float* Wkf = (const float*)d_in[11];
    const float* bkf = (const float*)d_in[12];
    __bf16* packed   = (__bf16*)d_ws;
    float*  out      = (float*)d_out;

    const long total = (long)NMAT * MAT_ELEMS;
    const int  pblocks = (int)((total + 255) / 256);
    pack_weights_kernel<<<pblocks, 256, 0, stream>>>(Wq, Wk, Wfw, Wfc, Wqf, Wkf, packed);

    const int smem_bytes = 147456 + 32768 + 32768 + 1024;   // ~214 KB < 320 KB WGP LDS
    const int batch = in_sizes[0] / (NPOS * DMODEL);
    pt_fused_kernel<<<batch / TILE_M, THREADS, smem_bytes, stream>>>(
        x, bq, bk, bfw, bfc, bqf, bkf, packed, out);
}
